// RealMambaSSM_33998961115595
// MI455X (gfx1250) — compile-verified, matrix operand order checked by reading, not run on hardware
//
#include <hip/hip_runtime.h>
#include <hip/hip_bf16.h>

typedef __attribute__((ext_vector_type(16))) _Float16 v16h;
typedef __attribute__((ext_vector_type(8)))  _Float16 v8h;
typedef __attribute__((ext_vector_type(8)))  float    v8f;
typedef __attribute__((ext_vector_type(4)))  int      i32x4;

// ---------------------------------------------------------------------------
// CDNA5 async global->LDS copy (16B per lane), with sync fallback.
// Builtin prototype (probe-confirmed by diagnostic): (int4 AS1*, int4 AS3*, Ii, Ii)
// ---------------------------------------------------------------------------
#if defined(__has_builtin)
#if __has_builtin(__builtin_amdgcn_global_load_async_to_lds_b128)
#define HAVE_ASYNC_LDS 1
#endif
#endif

typedef __attribute__((address_space(1))) i32x4* gas_i32x4_ptr;
typedef __attribute__((address_space(3))) i32x4* las_i32x4_ptr;

static __device__ __forceinline__ void copy16_to_lds(const _Float16* __restrict__ g,
                                                     _Float16* __restrict__ l) {
#ifdef HAVE_ASYNC_LDS
  __builtin_amdgcn_global_load_async_to_lds_b128(
      (gas_i32x4_ptr)(_Float16*)g, (las_i32x4_ptr)l, 0, 0);
#else
  *reinterpret_cast<v8h*>(l) = *reinterpret_cast<const v8h*>(g);
#endif
}

static __device__ __forceinline__ void wait_async_all() {
#ifdef HAVE_ASYNC_LDS
#if __has_builtin(__builtin_amdgcn_s_wait_asynccnt)
  __builtin_amdgcn_s_wait_asynccnt(0);
#else
  asm volatile("s_wait_asynccnt 0x0" ::: "memory");
#endif
#endif
}

// ---------------------------------------------------------------------------
// Fragment load from LDS tile (row-major, padded row = LDL halfs).
// WMMA 16-bit A/B layout (16x32): lanes 0-15 hold row M=lane, K={0..7,16..23};
// lanes 16-31 hold row M=lane-16, K={8..15,24..31}. (B stored [N x K]: same.)
// ---------------------------------------------------------------------------
static __device__ __forceinline__ v16h lds_frag(const _Float16* __restrict__ lds,
                                                int ldl, int row0, int lane) {
  const int r  = row0 + (lane & 15);
  const int kb = (lane >> 4) << 3;   // +0 or +8 halfs
  const _Float16* p = lds + r * ldl + kb;
  const v8h lo = *reinterpret_cast<const v8h*>(p);        // K = kb .. kb+7
  const v8h hi = *reinterpret_cast<const v8h*>(p + 16);   // K = kb+16 .. kb+23
  v16h out;
#pragma unroll
  for (int i = 0; i < 8; ++i) { out[i] = lo[i]; out[i + 8] = hi[i]; }
  return out;
}

// f32 C/D layout: lane 0-15 -> N=lane, VGPR r -> M=r; lanes 16-31 -> N=lane-16, M=r+8.
static __device__ __forceinline__ void store_tile_f32(float* __restrict__ C, int ldc,
                                                      int m0, int n0, const v8f& acc, int lane) {
  const int n  = n0 + (lane & 15);
  const int mb = m0 + ((lane >> 4) << 3);
#pragma unroll
  for (int r = 0; r < 8; ++r) C[(size_t)(mb + r) * ldc + n] = acc[r];
}

static __device__ __forceinline__ void store_tile_f16(_Float16* __restrict__ C, int ldc,
                                                      int m0, int n0, const v8f& acc, int lane) {
  const int n  = n0 + (lane & 15);
  const int mb = m0 + ((lane >> 4) << 3);
#pragma unroll
  for (int r = 0; r < 8; ++r) C[(size_t)(mb + r) * ldc + n] = (_Float16)acc[r];
}

// ---------------------------------------------------------------------------
// LDS-staged, double-buffered WMMA GEMM: C[M,N] = A[M,K] * B[N,K]^T.
// Block: 256 threads = 8 waves (wave32); block tile 128x128x32; wave tile 32x64.
// Tiles staged via async global->LDS (ASYNCcnt), fragments via ds_load_b128.
// SPLIT epilogue: cols < splitN -> f32 C0 (ld=splitN), else f16 C1 (ld=splitN).
// All problem dims are multiples of the tile sizes -> no bounds checks.
// ---------------------------------------------------------------------------
template <bool SPLIT>
__global__ __launch_bounds__(256) void wmma_gemm_kernel(
    const _Float16* __restrict__ A, const _Float16* __restrict__ Bm,
    float* __restrict__ C0, _Float16* __restrict__ C1,
    int M, int N, int K, int lda, int ldb, int ldc, int splitN) {
  constexpr int LDL = 40;   // 32 halfs + 8 pad -> 80B row stride: conflict-free b128 reads
  __shared__ _Float16 ldsA[2][128 * LDL];
  __shared__ _Float16 ldsB[2][128 * LDL];

  const int tid   = threadIdx.x;
  const int lane  = tid & 31;
  const int wave  = tid >> 5;
  const int waveM = wave >> 1;          // 0..3  (32 rows each)
  const int waveN = wave & 1;           // 0..1  (64 cols each)
  const int bm = blockIdx.y * 128;
  const int bn = blockIdx.x * 128;

  // Stage one 128x32 A tile + 128x32 B tile (16B chunks; 4 per thread).
  auto issue_tile = [&](int k0, int buf) {
#pragma unroll
    for (int i = 0; i < 2; ++i) {
      const int idx = tid + i * 256;
      const int row = idx >> 2;
      const int ch  = (idx & 3) * 8;                 // half offset within row
      copy16_to_lds(A  + (size_t)(bm + row) * lda + k0 + ch, &ldsA[buf][row * LDL + ch]);
      copy16_to_lds(Bm + (size_t)(bn + row) * ldb + k0 + ch, &ldsB[buf][row * LDL + ch]);
    }
  };

  v8f acc[2][4];
#pragma unroll
  for (int i = 0; i < 2; ++i)
#pragma unroll
    for (int j = 0; j < 4; ++j) acc[i][j] = {};

  const int nk = K >> 5;
  issue_tile(0, 0);
  for (int it = 0; it < nk; ++it) {
    wait_async_all();
    __syncthreads();                       // tile `it` visible to all waves
    if (it + 1 < nk) issue_tile((it + 1) << 5, (it + 1) & 1);  // overlap next copy

    const int buf = it & 1;
    v16h af[2], bf[4];
#pragma unroll
    for (int i = 0; i < 2; ++i) af[i] = lds_frag(&ldsA[buf][0], LDL, waveM * 32 + i * 16, lane);
#pragma unroll
    for (int j = 0; j < 4; ++j) bf[j] = lds_frag(&ldsB[buf][0], LDL, waveN * 64 + j * 16, lane);
#pragma unroll
    for (int i = 0; i < 2; ++i)
#pragma unroll
      for (int j = 0; j < 4; ++j)
        acc[i][j] = __builtin_amdgcn_wmma_f32_16x16x32_f16(
            false, af[i], false, bf[j], (short)0, acc[i][j], false, false);
    __syncthreads();                       // reads of tile `it` done before reuse
  }

  const int wm = bm + waveM * 32;
  const int wn = bn + waveN * 64;
  if (!SPLIT || (bn < splitN)) {
    const int ld = SPLIT ? splitN : ldc;
#pragma unroll
    for (int i = 0; i < 2; ++i)
#pragma unroll
      for (int j = 0; j < 4; ++j)
        store_tile_f32(C0, ld, wm + i * 16, wn + j * 16, acc[i][j], lane);
  } else {
#pragma unroll
    for (int i = 0; i < 2; ++i)
#pragma unroll
      for (int j = 0; j < 4; ++j)
        store_tile_f16(C1, splitN, wm + i * 16, wn - splitN + j * 16, acc[i][j], lane);
  }
}

// ---------------------------------------------------------------------------
// Elementwise / small kernels
// ---------------------------------------------------------------------------
__global__ void cvt_f32_f16_kernel(const float* __restrict__ in, _Float16* __restrict__ out, size_t n) {
  size_t i = (size_t)blockIdx.x * blockDim.x + threadIdx.x;
  const size_t stride = (size_t)gridDim.x * blockDim.x;
  for (; i < n; i += stride) out[i] = (_Float16)in[i];
}

// depthwise conv1d, k=3, pad=1, along T (cross-correlation, XLA semantics)
__global__ void conv3_kernel(const float* __restrict__ xs, const float* __restrict__ w,
                             const float* __restrict__ bias, float* __restrict__ xc,
                             int T, int C, size_t total) {
  const size_t idx = (size_t)blockIdx.x * blockDim.x + threadIdx.x;
  if (idx >= total) return;
  const int c = (int)(idx % C);
  const int t = (int)((idx / C) % T);
  const float x0 = (t > 0)     ? xs[idx - C] : 0.0f;
  const float x1 = xs[idx];
  const float x2 = (t < T - 1) ? xs[idx + C] : 0.0f;
  xc[idx] = w[c * 3 + 0] * x0 + w[c * 3 + 1] * x1 + w[c * 3 + 2] * x2 + bias[c];
}

// v[r] = mean_c dt_proj_w[c,r] ; bbar = mean(dt_proj_b)
__global__ void prep_v_kernel(const float* __restrict__ dtw, const float* __restrict__ dtb,
                              float* __restrict__ v, float* __restrict__ bbar, int Ci, int R) {
  const int r = threadIdx.x;
  if (r < R) {
    float s = 0.0f;
    for (int c = 0; c < Ci; ++c) s += dtw[(size_t)c * R + r];
    v[r] = s / (float)Ci;
  } else if (r == R) {
    float s = 0.0f;
    for (int c = 0; c < Ci; ++c) s += dtb[c];
    bbar[0] = s / (float)Ci;
  }
}

// u[c] = sum_r x_proj_w[r,c] * v[r]   (dt rows of x_proj_w are rows 0..R-1)
__global__ void prep_u_kernel(const float* __restrict__ xpw, const float* __restrict__ v,
                              float* __restrict__ u, int Ci, int R) {
  const int c = blockIdx.x * blockDim.x + threadIdx.x;
  if (c >= Ci) return;
  float s = 0.0f;
  for (int r = 0; r < R; ++r) s += xpw[(size_t)r * Ci + c] * v[r];
  u[c] = s;
}

// gate[m] = sigmoid(dot(xc[m,:], u) + bbar) ; one wave32 per token
__global__ void gate_kernel(const float* __restrict__ xc, const float* __restrict__ u,
                            const float* __restrict__ bbar, float* __restrict__ gate,
                            int tokens, int C) {
  const int wave = (int)(((size_t)blockIdx.x * blockDim.x + threadIdx.x) >> 5);
  const int lane = threadIdx.x & 31;
  if (wave >= tokens) return;
  const float* row = xc + (size_t)wave * C;
  float s = 0.0f;
  for (int i = lane; i < C; i += 32) s += row[i] * u[i];
#pragma unroll
  for (int off = 16; off > 0; off >>= 1) s += __shfl_xor(s, off, 32);
  if (lane == 0) gate[wave] = 1.0f / (1.0f + __expf(-(s + bbar[0])));
}

// ---------------------------------------------------------------------------
// Chunked scan:  h_t = g_t h_{t-1} + (1-g_t) x_t
// Phase 0: per-(b,chunk) gate product A_j.
// Phase 1: per-(b,chunk,c) local scan with h0=0 -> yloc (f16).
// Phase 2: per-(b,c) carry across chunks: hin[j+1] = A_j hin[j] + yloc[last].
// Phase 3: y_t = cum_t*hin + yloc_t ; then y = y*silu(z) + D*x ; write f16.
// ---------------------------------------------------------------------------
__global__ void scan_chunkprod_kernel(const float* __restrict__ gate, float* __restrict__ Aprod,
                                      int B, int T, int CH) {
  const int NCH = T / CH;
  const int idx = blockIdx.x * blockDim.x + threadIdx.x;
  if (idx >= B * NCH) return;
  const int b = idx / NCH, j = idx % NCH;
  const float* g = gate + (size_t)b * T + (size_t)j * CH;
  float p = 1.0f;
  for (int t = 0; t < CH; ++t) p *= g[t];
  Aprod[idx] = p;
}

__global__ void scan_local_kernel(const float* __restrict__ xc, const float* __restrict__ gate,
                                  _Float16* __restrict__ yloc, int B, int T, int C, int CH) {
  const int NCH = T / CH;
  const size_t idx = (size_t)blockIdx.x * blockDim.x + threadIdx.x;
  if (idx >= (size_t)B * NCH * C) return;
  const int c = (int)(idx % C);
  const int j = (int)((idx / C) % NCH);
  const int b = (int)(idx / ((size_t)C * NCH));
  const size_t base = ((size_t)b * T + (size_t)j * CH) * C + c;
  const float* g = gate + (size_t)b * T + (size_t)j * CH;
  float h = 0.0f;
  for (int t = 0; t < CH; ++t) {
    const float gv = g[t];
    h = gv * h + (1.0f - gv) * xc[base + (size_t)t * C];
    yloc[base + (size_t)t * C] = (_Float16)h;
  }
}

__global__ void scan_carry_kernel(const float* __restrict__ Aprod, const _Float16* __restrict__ yloc,
                                  float* __restrict__ hin, int B, int T, int C, int CH) {
  const int NCH = T / CH;
  const int idx = blockIdx.x * blockDim.x + threadIdx.x;
  if (idx >= B * C) return;
  const int c = idx % C, b = idx / C;
  float h = 0.0f;
  for (int j = 0; j < NCH; ++j) {
    hin[((size_t)b * NCH + j) * C + c] = h;
    const float hl = (float)yloc[((size_t)b * T + (size_t)j * CH + (CH - 1)) * C + c];
    h = Aprod[b * NCH + j] * h + hl;
  }
}

__global__ void scan_final_kernel(const float* __restrict__ hin, const float* __restrict__ gate,
                                  const _Float16* __restrict__ zh, const float* __restrict__ xc,
                                  const float* __restrict__ Dp, _Float16* __restrict__ y,
                                  int B, int T, int C, int CH) {
  const int NCH = T / CH;
  const size_t idx = (size_t)blockIdx.x * blockDim.x + threadIdx.x;
  if (idx >= (size_t)B * NCH * C) return;
  const int c = (int)(idx % C);
  const int j = (int)((idx / C) % NCH);
  const int b = (int)(idx / ((size_t)C * NCH));
  const size_t base = ((size_t)b * T + (size_t)j * CH) * C + c;
  const float* g = gate + (size_t)b * T + (size_t)j * CH;
  const float h0 = hin[((size_t)b * NCH + j) * C + c];
  const float dv = Dp[c];
  float cum = 1.0f;
  for (int t = 0; t < CH; ++t) {
    cum *= g[t];
    const size_t id = base + (size_t)t * C;
    float yv = cum * h0 + (float)y[id];          // fix up local scan (in place)
    const float zv = (float)zh[id];
    const float sz = zv / (1.0f + __expf(-zv));  // silu(z)
    yv = yv * sz + dv * xc[id];
    y[id] = (_Float16)yv;
  }
}

// ---------------------------------------------------------------------------
extern "C" void kernel_launch(void* const* d_in, const int* in_sizes, int n_in,
                              void* d_out, int out_size, void* d_ws, size_t ws_size,
                              hipStream_t stream) {
  (void)in_sizes; (void)n_in; (void)out_size; (void)ws_size;
  const float* x     = (const float*)d_in[0];   // [4,4096,1024]
  const float* w_in  = (const float*)d_in[1];   // [4096,1024]
  const float* convw = (const float*)d_in[2];   // [2048,1,3]
  const float* convb = (const float*)d_in[3];   // [2048]
  const float* xpw   = (const float*)d_in[4];   // [96,2048]
  const float* dtw   = (const float*)d_in[5];   // [2048,64]
  const float* dtb   = (const float*)d_in[6];   // [2048]
  const float* Dp    = (const float*)d_in[7];   // [2048]
  const float* wout  = (const float*)d_in[8];   // [1024,2048]
  float* out = (float*)d_out;                   // [4,4096,1024]

  constexpr int Bb = 4, T = 4096, Dm = 1024, Ci = 2048, R = 64, CH = 64;
  constexpr int NCH = T / CH;
  constexpr int M  = Bb * T;     // 16384 tokens
  constexpr int N1 = 2 * Ci;     // 4096

  // Workspace carve-out (256B aligned)
  char* p = (char*)d_ws;
  auto carve = [&](size_t bytes) -> char* {
    char* q = p; p += (bytes + 255) & ~(size_t)255; return q;
  };
  _Float16* xh    = (_Float16*)carve((size_t)M * Dm * 2);    //  32 MB
  _Float16* w1h   = (_Float16*)carve((size_t)N1 * Dm * 2);   //   8 MB
  _Float16* wouth = (_Float16*)carve((size_t)Dm * Ci * 2);   //   4 MB
  float*    xs    = (float*)   carve((size_t)M * Ci * 4);    // 128 MB (pre-conv)
  _Float16* zh    = (_Float16*)carve((size_t)M * Ci * 2);    //  64 MB (gate branch)
  float*    xc    = (float*)   carve((size_t)M * Ci * 4);    // 128 MB (post-conv)
  _Float16* yh    = (_Float16*)carve((size_t)M * Ci * 2);    //  64 MB (scan/combined)
  float*    gate  = (float*)   carve((size_t)M * 4);
  float*    u     = (float*)   carve((size_t)Ci * 4);
  float*    vv    = (float*)   carve((size_t)R * 4);
  float*    bbar  = (float*)   carve(4);
  float*    Aprod = (float*)   carve((size_t)Bb * NCH * 4);
  float*    hin   = (float*)   carve((size_t)Bb * NCH * Ci * 4);

  const dim3 blk(256);

  // 1) f32 -> f16 conversions (activations + GEMM weights)
  cvt_f32_f16_kernel<<<dim3(4096), blk, 0, stream>>>(x,    xh,    (size_t)M * Dm);
  cvt_f32_f16_kernel<<<dim3(2048), blk, 0, stream>>>(w_in, w1h,   (size_t)N1 * Dm);
  cvt_f32_f16_kernel<<<dim3(1024), blk, 0, stream>>>(wout, wouth, (size_t)Dm * Ci);

  // 2) gate weight collapse: v,bbar then u
  prep_v_kernel<<<dim3(1), dim3(128), 0, stream>>>(dtw, dtb, vv, bbar, Ci, R);
  prep_u_kernel<<<dim3(Ci / 256), blk, 0, stream>>>(xpw, vv, u, Ci, R);

  // 3) in_proj GEMM: [M,2*Ci] = xh * w1h^T, split -> xs (f32) | zh (f16)
  wmma_gemm_kernel<true><<<dim3(N1 / 128, M / 128), blk, 0, stream>>>(
      xh, w1h, xs, zh, M, N1, Dm, Dm, Dm, /*ldc unused*/ Ci, /*splitN*/ Ci);

  // 4) depthwise conv1d k=3 pad=1 along T
  {
    const size_t total = (size_t)M * Ci;
    conv3_kernel<<<dim3((unsigned)((total + 255) / 256)), blk, 0, stream>>>(
        xs, convw, convb, xc, T, Ci, total);
  }

  // 5) gate[m] = sigmoid(dot(xc[m,:], u) + bbar)
  gate_kernel<<<dim3((M * 32) / 256), blk, 0, stream>>>(xc, u, bbar, gate, M, Ci);

  // 6) chunked selective scan + SiLU-gate combine -> yh (f16)
  scan_chunkprod_kernel<<<dim3(1), blk, 0, stream>>>(gate, Aprod, Bb, T, CH);
  {
    const size_t total = (size_t)Bb * NCH * Ci;
    scan_local_kernel<<<dim3((unsigned)((total + 255) / 256)), blk, 0, stream>>>(
        xc, gate, yh, Bb, T, Ci, CH);
    scan_carry_kernel<<<dim3((Bb * Ci) / 256), blk, 0, stream>>>(Aprod, yh, hin, Bb, T, Ci, CH);
    scan_final_kernel<<<dim3((unsigned)((total + 255) / 256)), blk, 0, stream>>>(
        hin, gate, zh, xc, Dp, yh, Bb, T, Ci, CH);
  }

  // 7) out_proj GEMM: out[M,Dm] = yh * wouth^T (f32 output)
  wmma_gemm_kernel<false><<<dim3(Dm / 128, M / 128), blk, 0, stream>>>(
      yh, wouth, out, nullptr, M, Dm, Ci, Ci, Ci, Dm, 0);
}
